// KeyformerAttn_75488345194582
// MI455X (gfx1250) — matrix operand order repaired and checked by reading, not compiled, not run on hardware
//
#include <hip/hip_runtime.h>

// Problem dims (compile-time)
#define Msz 256
#define Hsz 32
#define Dsz 128
#define Psz 4096
#define Nsz 4096   // H*D
#define Lsz 4352   // P + M
#define NCHUNK (Lsz / 64)

typedef __attribute__((ext_vector_type(16))) __bf16 v16bf;
typedef __attribute__((ext_vector_type(8)))  float  v8f;
typedef __attribute__((ext_vector_type(4)))  unsigned int v4u;
typedef __attribute__((ext_vector_type(8)))  int  v8i;
typedef __attribute__((ext_vector_type(4)))  int  v4i;

__device__ inline __bf16 f2bf(float x) {
  return (__bf16)x;   // native v_cvt_pk_bf16_f32 path (RNE)
}

__device__ inline v8f vzero8() {
  v8f z;
#pragma unroll
  for (int i = 0; i < 8; ++i) z[i] = 0.0f;
  return z;
}

__device__ inline v8f wmma_bf16(v16bf a, v16bf b, v8f c) {
  // 8 args: (neg_a, A, neg_b, B, c_mod, C, reuse_a, reuse_b)
  return __builtin_amdgcn_wmma_f32_16x16x32_bf16(false, a, false, b, (short)0, c,
                                                 false, false);
}

// ---- Tensor Data Mover: 64x128 bf16 tile, global(contig) -> LDS(pitch 136) ----
// D# per CDNA5 ISA ch.8: group0 = {flags/count, lds_addr, global_addr, type=2},
// group1 = {mask/size/pad, dims, strides}. Groups 2/3 zero (2-D tensor).
// Padding: after every 64 DWORDs (one 256B row) insert 4 DWORDs (8 bf16)
// -> LDS row pitch 136 bf16, matching the bank-conflict-free tile layout.
__device__ inline void tdm_load_tile_64x128(const __bf16* gsrc, __bf16* lds) {
  unsigned lds_off = (unsigned)(size_t)lds;               // addr[31:0] = LDS byte offset
  unsigned long long ga = (unsigned long long)(size_t)gsrc;
  v4u g0;
  g0[0] = 1u;                                             // count=1 (valid), user mode
  g0[1] = lds_off;                                        // lds_addr
  g0[2] = (unsigned)(ga & 0xFFFFFFFFu);                   // global_addr[31:0]
  g0[3] = (unsigned)((ga >> 32) & 0x01FFFFFFu) | (2u << 30); // addr[56:32] | type=2
  v8i g1;
  g1[0] = (int)((1u << 16)      // data_size = 1 -> 2 bytes
              | (1u << 20)      // pad_enable
              | (5u << 22)      // pad_interval = 64 DWORDs (one row)
              | (3u << 25));    // pad_amount  = 4 DWORDs (8 bf16)
  g1[1] = (int)(128u << 16);    // atomic_addr=0 | tensor_dim0[15:0] = 128
  g1[2] = (int)(64u << 16);     // tensor_dim0[31:16]=0 | tensor_dim1[15:0] = 64
  g1[3] = (int)(128u << 16);    // tensor_dim1[31:16]=0 | tile_dim0 = 128
  g1[4] = (int)(64u);           // tile_dim1 = 64 | tile_dim2 = 0
  g1[5] = (int)(128);           // tensor_dim0_stride[31:0] = 128 elements
  g1[6] = 0;                    // stride0[47:32]=0 | stride1[15:0]=0
  g1[7] = 0;
  v4i g2 = {0, 0, 0, 0};
  v4i g3 = {0, 0, 0, 0};
  v8i gx = {0, 0, 0, 0, 0, 0, 0, 0};   // extra group (6-arg toolchain variant)
  __builtin_amdgcn_tensor_load_to_lds(g0, g1, g2, g3, gx, 0);
}

// ---- WMMA fragment loaders (wave32; this code's contiguous k = 16*(lane/16) + s, same for A and B) ----------
// A (16x32, 16-bit): lanes 0-15 row m=lane, K=0..15; lanes 16-31 row
// m=lane-16, K=16..31. Two bf16 per VGPR.
__device__ inline v16bf load_a_frag(const __bf16* lds, int pitch, int row0, int k0,
                                    int lane) {
  int m = row0 + (lane & 15);
  int kbase = k0 + ((lane & 16) ? 16 : 0);
  const unsigned* p = (const unsigned*)(lds + m * pitch);  // pitch is even
  int kb2 = kbase >> 1;
  union { v16bf v; unsigned u[8]; } r;
#pragma unroll
  for (int j = 0; j < 4; ++j) {
    r.u[j]     = p[kb2 + j];       // K = kbase + 2j, +1
    r.u[4 + j] = p[kb2 + 4 + j];   // K = kbase + 8 + 2j, +1
  }
  return r.v;
}

// B (32x16, 16-bit): lanes 0-15 col n=lane K=0..15; lanes 16-31 col n=lane-16
// K=16..31. LDS stored n-major (row n holds contiguous K): contiguous reads.
__device__ inline v16bf load_b_frag_nmajor(const __bf16* lds, int pitch, int n0,
                                           int k0, int lane) {
  int n = n0 + (lane & 15);
  int kbase = k0 + ((lane & 16) ? 16 : 0);
  const unsigned* p = (const unsigned*)(lds + n * pitch + kbase);
  union { v16bf v; unsigned u[8]; } r;
#pragma unroll
  for (int j = 0; j < 8; ++j) r.u[j] = p[j];
  return r.v;
}

// Same B layout but LDS is stored k-major (row k holds contiguous n): strided.
__device__ inline v16bf load_b_frag_kmajor(const __bf16* lds, int pitch, int k0,
                                           int n0, int lane) {
  int n = n0 + (lane & 15);
  int kbase = k0 + ((lane & 16) ? 16 : 0);
  union { v16bf v; __bf16 b[16]; } r;
#pragma unroll
  for (int i = 0; i < 16; ++i) r.b[i] = lds[(kbase + i) * pitch + n];
  return r.v;
}

// ---------------------------------------------------------------------------
// Kernel 0: convert f32 caches into bf16 [H, L, D] combined cache rows [0,P)
__global__ __launch_bounds__(256) void conv_cache_kernel(
    const float* __restrict__ cK, const float* __restrict__ cV,
    __bf16* __restrict__ kc, __bf16* __restrict__ vc) {
  const float* src = blockIdx.y ? cV : cK;
  __bf16* dst = blockIdx.y ? vc : kc;
  size_t i = (size_t)blockIdx.x * 256 + threadIdx.x;     // float4 index
  float4 v = ((const float4*)src)[i];
  size_t e = i * 4;                                      // element index in [H,P,D]
  size_t h = e >> 19;                                    // P*D = 2^19
  size_t rem = e & ((1u << 19) - 1);
  __bf16* d = dst + h * ((size_t)Lsz * Dsz) + rem;
  d[0] = f2bf(v.x); d[1] = f2bf(v.y); d[2] = f2bf(v.z); d[3] = f2bf(v.w);
}

// ---------------------------------------------------------------------------
// Kernel 1: QKV projection. C[M,N] = X @ W^T. Block = 4 waves, tile 256x16.
// blockIdx.x: N-tile (0..255). blockIdx.y: 0=q, 1=k, 2=v.
__global__ __launch_bounds__(128) void qkv_proj_kernel(
    const float* __restrict__ X, const float* __restrict__ Wq,
    const float* __restrict__ Wk, const float* __restrict__ Wv,
    __bf16* __restrict__ qbf, __bf16* __restrict__ kcbf, __bf16* __restrict__ vcbf) {
  const int tid = threadIdx.x, lane = tid & 31, wave = tid >> 5;
  const int which = blockIdx.y;
  const float* W = (which == 0) ? Wq : (which == 1) ? Wk : Wv;
  const int n0 = blockIdx.x * 16;

  __shared__ __bf16 xa[256 * 40];   // X tile  [256 rows][32 K], pitch 40
  __shared__ __bf16 wt[16 * 40];    // W tile  [16 n-rows][32 K], pitch 40

  v8f acc[4];
#pragma unroll
  for (int s = 0; s < 4; ++s) acc[s] = vzero8();

  for (int k0 = 0; k0 < Nsz; k0 += 32) {
    // X chunk: 256x32 f32 -> bf16 LDS (2048 float4 loads / 128 threads)
    for (int idx = tid; idx < 256 * 8; idx += 128) {
      int row = idx >> 3, c4 = idx & 7;
      float4 v = ((const float4*)(X + (size_t)row * Nsz + k0))[c4];
      __bf16* d = &xa[row * 40 + c4 * 4];
      d[0] = f2bf(v.x); d[1] = f2bf(v.y); d[2] = f2bf(v.z); d[3] = f2bf(v.w);
    }
    // W chunk: 16x32 f32 -> bf16 LDS (128 float4 loads)
    {
      int row = tid >> 3, c4 = tid & 7;
      float4 v = ((const float4*)(W + (size_t)(n0 + row) * Nsz + k0))[c4];
      __bf16* d = &wt[row * 40 + c4 * 4];
      d[0] = f2bf(v.x); d[1] = f2bf(v.y); d[2] = f2bf(v.z); d[3] = f2bf(v.w);
    }
    if (k0 + 32 < Nsz)   // pull next W chunk toward L2 (global_prefetch_b8)
      __builtin_prefetch(W + (size_t)(n0 + (tid >> 3)) * Nsz + k0 + 32, 0, 1);
    __syncthreads();

    v16bf b = load_b_frag_nmajor(wt, 40, 0, 0, lane);
#pragma unroll
    for (int s = 0; s < 4; ++s) {
      v16bf a = load_a_frag(xa, 40, (wave + 4 * s) * 16, 0, lane);
      acc[s] = wmma_bf16(a, b, acc[s]);
    }
    __syncthreads();
  }

  // Write out per C layout: col n = lane&15, rows r (+8 for upper lane half)
  const int n = n0 + (lane & 15);
  const int h = n >> 7, d = n & 127;
  const int base = (lane & 16) ? 8 : 0;
#pragma unroll
  for (int s = 0; s < 4; ++s) {
#pragma unroll
    for (int r = 0; r < 8; ++r) {
      int m = (wave + 4 * s) * 16 + base + r;
      __bf16 val = f2bf(acc[s][r]);
      if (which == 0)      qbf [((size_t)h * Msz + m) * Dsz + d] = val;
      else if (which == 1) kcbf[((size_t)h * Lsz + Psz + m) * Dsz + d] = val;
      else                 vcbf[((size_t)h * Lsz + Psz + m) * Dsz + d] = val;
    }
  }
}

// ---------------------------------------------------------------------------
// Kernel 2 (pass A): per (m-tile, head) compute row sums of exp(c) and
// exp((c+noise)/tau) over full L. Block = 4 waves; wave w owns score cols
// [w*16, w*16+16) of each 64-wide L chunk.
// K tiles stream via TDM, double-buffered: wave 0 issues chunk i+1's DMA,
// then waits tensorcnt<=1 (in-order completion => chunk i resident).
__global__ __launch_bounds__(128) void attn_rowsum_kernel(
    const __bf16* __restrict__ qbf, const __bf16* __restrict__ kcbf,
    const float* __restrict__ noise, const float* __restrict__ tau,
    float* __restrict__ rs_e, float* __restrict__ rs_ep) {
  const int tid = threadIdx.x, lane = tid & 31, wave = tid >> 5;
  const int mt = blockIdx.x, h = blockIdx.y;

  __shared__ __bf16 qa[16 * 136];
  __shared__ __bf16 kt[2][64 * 136];
  __shared__ float red[32];

  const float inv_tau = 1.0f / tau[0];
  const __bf16* kbase_g = kcbf + (size_t)h * Lsz * Dsz;

  // q tile [16][128] bf16 -> LDS (b32 copies; pitch 136 bf16 = 68 u32)
  {
    const unsigned* qs = (const unsigned*)(qbf + ((size_t)h * Msz + mt * 16) * Dsz);
    unsigned* qau = (unsigned*)qa;
    for (int idx = tid; idx < 16 * 64; idx += 128)
      qau[(idx >> 6) * 68 + (idx & 63)] = qs[idx];
  }
  if (tid < 32) red[tid] = 0.0f;

  float ae[8], aep[8];
#pragma unroll
  for (int r = 0; r < 8; ++r) { ae[r] = 0.0f; aep[r] = 0.0f; }

  const int base = (lane & 16) ? 8 : 0;
  const int mrow = mt * 16 + base;

  if (wave == 0) tdm_load_tile_64x128(kbase_g, kt[0]);   // prologue: chunk 0

  for (int ci = 0; ci < NCHUNK; ++ci) {
    const int buf = ci & 1;
    if (wave == 0) {
      if (ci + 1 < NCHUNK) {
        tdm_load_tile_64x128(kbase_g + (size_t)(ci + 1) * 64 * Dsz, kt[buf ^ 1]);
        __builtin_amdgcn_s_wait_tensorcnt(1);   // chunk ci done (in-order)
      } else {
        __builtin_amdgcn_s_wait_tensorcnt(0);
      }
    }
    __syncthreads();

    v8f s = vzero8();
#pragma unroll
    for (int kk = 0; kk < 4; ++kk) {
      v16bf a = load_a_frag(qa, 136, 0, kk * 32, lane);
      v16bf b = load_b_frag_nmajor(kt[buf], 136, wave * 16, kk * 32, lane);
      s = wmma_bf16(a, b, s);
    }

    const int l = ci * 64 + wave * 16 + (lane & 15);
    const float* nzp = noise + ((size_t)h * Msz + mrow) * Lsz + l;
#pragma unroll
    for (int r = 0; r < 8; ++r) {
      float sc = s[r];
      float nz = nzp[(size_t)r * Lsz];
      ae[r]  += __expf(sc);
      aep[r] += __expf((sc + nz) * inv_tau);
    }
    __syncthreads();   // all reads of kt[buf] done before its next DMA overwrite
  }

#pragma unroll
  for (int r = 0; r < 8; ++r) {
    atomicAdd(&red[base + r], ae[r]);
    atomicAdd(&red[16 + base + r], aep[r]);
  }
  __syncthreads();
  if (tid < 16) {
    rs_e [h * Msz + mt * 16 + tid] = red[tid];
    rs_ep[h * Msz + mt * 16 + tid] = red[16 + tid];
  }
}

// ---------------------------------------------------------------------------
__global__ __launch_bounds__(256) void zero_f32_kernel(float* __restrict__ p, int n) {
  int i = blockIdx.x * 256 + threadIdx.x;
  if (i < n) p[i] = 0.0f;
}

// ---------------------------------------------------------------------------
// Kernel 3 (pass B): recompute scores, normalize with pass-A sums, accumulate
// token scores (c_out) atomically, accumulate O = P @ Vc with WMMA.
// K and V tiles stream via TDM, double-buffered (2 descriptors per chunk;
// wait tensorcnt<=2 after issuing the next pair => current pair resident).
__global__ __launch_bounds__(128) void attn_out_kernel(
    const __bf16* __restrict__ qbf, const __bf16* __restrict__ kcbf,
    const __bf16* __restrict__ vcbf, const float* __restrict__ noise,
    const float* __restrict__ tau, const float* __restrict__ rs_e,
    const float* __restrict__ rs_ep, float* __restrict__ out_o,
    float* __restrict__ out_c) {
  const int tid = threadIdx.x, lane = tid & 31, wave = tid >> 5;
  const int mt = blockIdx.x, h = blockIdx.y;

  __shared__ __bf16 qa[16 * 136];
  __shared__ __bf16 kt[2][64 * 136];
  __shared__ __bf16 vt[2][64 * 136];
  __shared__ __bf16 pt[16 * 72];
  __shared__ float rse_s[16], rsep_s[16];

  const float inv_tau = 1.0f / tau[0];
  const __bf16* kbase_g = kcbf + (size_t)h * Lsz * Dsz;
  const __bf16* vbase_g = vcbf + (size_t)h * Lsz * Dsz;

  {
    const unsigned* qs = (const unsigned*)(qbf + ((size_t)h * Msz + mt * 16) * Dsz);
    unsigned* qau = (unsigned*)qa;
    for (int idx = tid; idx < 16 * 64; idx += 128)
      qau[(idx >> 6) * 68 + (idx & 63)] = qs[idx];
  }
  if (tid < 16) {
    rse_s[tid]  = 1.0f / rs_e [h * Msz + mt * 16 + tid];
    rsep_s[tid] = 1.0f / rs_ep[h * Msz + mt * 16 + tid];
  }

  v8f oacc[2];
  oacc[0] = vzero8(); oacc[1] = vzero8();

  const int base = (lane & 16) ? 8 : 0;
  const int mrow = mt * 16 + base;

  if (wave == 0) {   // prologue: chunk 0
    tdm_load_tile_64x128(kbase_g, kt[0]);
    tdm_load_tile_64x128(vbase_g, vt[0]);
  }

  for (int ci = 0; ci < NCHUNK; ++ci) {
    const int buf = ci & 1;
    if (wave == 0) {
      if (ci + 1 < NCHUNK) {
        tdm_load_tile_64x128(kbase_g + (size_t)(ci + 1) * 64 * Dsz, kt[buf ^ 1]);
        tdm_load_tile_64x128(vbase_g + (size_t)(ci + 1) * 64 * Dsz, vt[buf ^ 1]);
        __builtin_amdgcn_s_wait_tensorcnt(2);   // chunk ci's pair done (in-order)
      } else {
        __builtin_amdgcn_s_wait_tensorcnt(0);
      }
    }
    __syncthreads();

    // Scores: S[16, 16] per wave, K = 128
    v8f s = vzero8();
#pragma unroll
    for (int kk = 0; kk < 4; ++kk) {
      v16bf a = load_a_frag(qa, 136, 0, kk * 32, lane);
      v16bf b = load_b_frag_nmajor(kt[buf], 136, wave * 16, kk * 32, lane);
      s = wmma_bf16(a, b, s);
    }

    const int l = ci * 64 + wave * 16 + (lane & 15);
    const float* nzp = noise + ((size_t)h * Msz + mrow) * Lsz + l;
    float sum_pp = 0.0f;
#pragma unroll
    for (int r = 0; r < 8; ++r) {
      float sc = s[r];
      float e  = __expf(sc);
      float ep = __expf((sc + nzp[(size_t)r * Lsz]) * inv_tau);
      float p  = e * rse_s[base + r];
      sum_pp  += ep * rsep_s[base + r];
      pt[(base + r) * 72 + wave * 16 + (lane & 15)] = f2bf(p);
    }
    atomicAdd(&out_c[(size_t)h * Lsz + l], sum_pp);
    __syncthreads();

    // O += P[16,64] @ V[64,128]; wave owns D cols [wave*32, wave*32+32)
#pragma unroll
    for (int kc = 0; kc < 2; ++kc) {
      v16bf a = load_a_frag(pt, 72, 0, kc * 32, lane);
#pragma unroll
      for (int t = 0; t < 2; ++t) {
        v16bf b = load_b_frag_kmajor(vt[buf], 136, kc * 32, wave * 32 + t * 16, lane);
        oacc[t] = wmma_bf16(a, b, oacc[t]);
      }
    }
    __syncthreads();   // all reads of kt/vt[buf] + pt done before reuse
  }

#pragma unroll
  for (int t = 0; t < 2; ++t) {
    int d = wave * 32 + t * 16 + (lane & 15);
#pragma unroll
    for (int r = 0; r < 8; ++r) {
      int m = mt * 16 + base + r;
      out_o[(size_t)m * Nsz + h * Dsz + d] = oacc[t][r];
    }
  }
}

// ---------------------------------------------------------------------------
extern "C" void kernel_launch(void* const* d_in, const int* in_sizes, int n_in,
                              void* d_out, int out_size, void* d_ws, size_t ws_size,
                              hipStream_t stream) {
  const float* X     = (const float*)d_in[0];
  const float* Wq    = (const float*)d_in[1];
  const float* Wk    = (const float*)d_in[2];
  const float* Wv    = (const float*)d_in[3];
  const float* cK    = (const float*)d_in[4];
  const float* cV    = (const float*)d_in[5];
  const float* tau   = (const float*)d_in[6];
  const float* noise = (const float*)d_in[7];

  float* out_o = (float*)d_out;
  float* out_c = out_o + (size_t)Msz * Nsz;   // [H, L] token scores

  // Workspace layout
  const size_t qbytes  = (size_t)Hsz * Msz * Dsz * 2;        // 2 MB
  const size_t kvbytes = (size_t)Hsz * Lsz * Dsz * 2;        // 35.65 MB each
  char* ws = (char*)d_ws;
  __bf16* qbf  = (__bf16*)ws;
  __bf16* kcbf = (__bf16*)(ws + qbytes);
  __bf16* vcbf = (__bf16*)(ws + qbytes + kvbytes);
  float*  rs_e  = (float*)(ws + qbytes + 2 * kvbytes);
  float*  rs_ep = rs_e + Hsz * Msz;

  // 0) cache f32 -> bf16 [H, L, D] rows [0, P)
  conv_cache_kernel<<<dim3((Hsz * Psz * Dsz) / (256 * 4), 2), 256, 0, stream>>>(
      cK, cV, kcbf, vcbf);

  // 1) QKV projections (q -> qbf; k,v -> cache rows [P, P+M))
  qkv_proj_kernel<<<dim3(Nsz / 16, 3), 128, 0, stream>>>(X, Wq, Wk, Wv, qbf, kcbf, vcbf);

  // 2) pass A: row sums over full L
  attn_rowsum_kernel<<<dim3(Msz / 16, Hsz), 128, 0, stream>>>(qbf, kcbf, noise, tau,
                                                              rs_e, rs_ep);

  // 3) zero token-score output (atomic accumulation target)
  zero_f32_kernel<<<(Hsz * Lsz + 255) / 256, 256, 0, stream>>>(out_c, Hsz * Lsz);

  // 4) pass B: normalize, token scores, O = P @ V
  attn_out_kernel<<<dim3(Msz / 16, Hsz), 128, 0, stream>>>(
      qbf, kcbf, vcbf, noise, tau, rs_e, rs_ep, out_o, out_c);
}
